// EnhancedGAT_32839319945833
// MI455X (gfx1250) — matrix operand order repaired and checked by reading, not compiled, not run on hardware
//
#include <hip/hip_runtime.h>
#include <hip/hip_bf16.h>

// ---------------- problem constants ----------------
#define NODES   50000
#define EDGES   600000
#define EPRIME  (EDGES + NODES)   // edges + self loops
#define F_IN    92
#define FK      96                // F_IN padded to mult of 32 for WMMA K
#define D1      256               // HEADS*HID
#define HIDC    32
#define NHEADS  8
#define NGRAPH  512
#define MPAD    50048             // NODES padded to mult of 128 (block row tile)
#define EPSBN   1e-5f

typedef __bf16 bf16;
typedef bf16  v16bf __attribute__((ext_vector_type(16)));
typedef float v8f   __attribute__((ext_vector_type(8)));

union FragAB { v16bf v; uint4 q[2]; };

__device__ inline uint4 ld16(const bf16* p) { return *reinterpret_cast<const uint4*>(p); }

// ---------------- precision / layout converters ----------------
__global__ void k_convx(const float* __restrict__ x, bf16* __restrict__ x16) {
    int t = blockIdx.x * 256 + threadIdx.x;
    if (t >= NODES * F_IN) return;
    int n = t / F_IN, k = t % F_IN;
    x16[(size_t)n * FK + k] = (bf16)x[t];
}
__global__ void k_convw1(const float* __restrict__ W1, bf16* __restrict__ w1t) {
    int t = blockIdx.x * 256 + threadIdx.x;
    if (t >= F_IN * D1) return;
    int k = t / D1, n = t % D1;          // W1 is [92,256] row-major
    w1t[(size_t)n * FK + k] = (bf16)W1[t];   // -> [256][96] K-contiguous
}
__global__ void k_convw2(const float* __restrict__ W2, bf16* __restrict__ w2t) {
    int t = blockIdx.x * 256 + threadIdx.x;
    if (t >= D1 * HIDC) return;
    int k = t / HIDC, n = t % HIDC;      // W2 is [256,32] row-major
    w2t[(size_t)n * D1 + k] = (bf16)W2[t];   // -> [32][256] K-contiguous
}
// ce[hd] = sum_c We1[hd*32+c]*ae1[hd*32+c]; ce[8] = sum_c We2[c]*ae2[c]
__global__ void k_ce(const float* __restrict__ We1, const float* __restrict__ ae1,
                     const float* __restrict__ We2, const float* __restrict__ ae2,
                     float* __restrict__ ce) {
    int t = threadIdx.x;
    if (t < NHEADS) {
        float s = 0.f;
        for (int c = 0; c < HIDC; ++c) s += We1[t * HIDC + c] * ae1[t * HIDC + c];
        ce[t] = s;
    } else if (t == NHEADS) {
        float s = 0.f;
        for (int c = 0; c < HIDC; ++c) s += We2[c] * ae2[c];
        ce[NHEADS] = s;
    }
}

// ---------------- self-loop attr = mean of incoming edge_attr ----------------
__global__ void k_loopA(const int* __restrict__ ei, const float* __restrict__ eattr,
                        float* __restrict__ cntin, float* __restrict__ asum) {
    int t = blockIdx.x * 256 + threadIdx.x;
    if (t >= EDGES) return;
    int d = ei[EDGES + t];
    atomicAdd(&cntin[d], 1.0f);
    atomicAdd(&asum[d], eattr[t]);
}
__global__ void k_loopB(const float* __restrict__ cntin, const float* __restrict__ asum,
                        float* __restrict__ lattr) {
    int t = blockIdx.x * 256 + threadIdx.x;
    if (t >= NODES) return;
    lattr[t] = asum[t] / fmaxf(cntin[t], 1.0f);
}

// ---------------- WMMA GEMM 1: [MPAD x 96]bf16 @ [96 x 256]bf16 -> f32 ----------------
// wave: 16 rows x 64 cols (4 wmma tiles); block: 8 waves = 128 rows; grid (MPAD/128, 4)
__global__ __launch_bounds__(256) void k_gemm1(const bf16* __restrict__ A,
                                               const bf16* __restrict__ Bt,
                                               float* __restrict__ C) {
    const int lane = threadIdx.x & 31;
    const int wid  = threadIdx.x >> 5;
    const int row0 = blockIdx.x * 128 + wid * 16;
    const int col0 = blockIdx.y * 64;
    const int mA    = lane & 15;
    const int koffA = (lane >> 4) << 3;   // A runs: k = koffA..+7 and 16+koffA..+7
    const int nB    = lane & 15;
    const int koffB = (lane >> 4) << 3;   // B runs: k = koffB..+7 and 16+koffB..+7
    v8f acc[4];
#pragma unroll
    for (int t = 0; t < 4; ++t)
#pragma unroll
        for (int j = 0; j < 8; ++j) acc[t][j] = 0.0f;

#pragma unroll
    for (int kb = 0; kb < FK; kb += 32) {
        FragAB a;
        const bf16* pa = A + (size_t)(row0 + mA) * FK + kb + koffA;
        a.q[0] = ld16(pa);
        a.q[1] = ld16(pa + 16);
        __builtin_prefetch(pa + 32, 0, 0);
#pragma unroll
        for (int t = 0; t < 4; ++t) {
            FragAB b;
            const bf16* pb = Bt + (size_t)(col0 + t * 16 + nB) * FK + kb + koffB;
            b.q[0] = ld16(pb);
            b.q[1] = ld16(pb + 16);
            acc[t] = __builtin_amdgcn_wmma_f32_16x16x32_bf16(
                false, a.v, false, b.v, (short)0, acc[t], false, false);
        }
    }
    const int rbase = row0 + ((lane >> 4) << 3);
    const int nc    = lane & 15;
#pragma unroll
    for (int t = 0; t < 4; ++t)
#pragma unroll
        for (int j = 0; j < 8; ++j)
            C[(size_t)(rbase + j) * D1 + col0 + t * 16 + nc] = acc[t][j];
}

// ---------------- WMMA GEMM 2: [MPAD x 256]bf16 @ [256 x 32]bf16 -> f32 ----------------
__global__ __launch_bounds__(256) void k_gemm2(const bf16* __restrict__ A,
                                               const bf16* __restrict__ Bt,
                                               float* __restrict__ C) {
    const int lane = threadIdx.x & 31;
    const int wid  = threadIdx.x >> 5;
    const int row0 = blockIdx.x * 128 + wid * 16;
    const int mA    = lane & 15;
    const int koffA = (lane >> 4) << 3;
    const int nB    = lane & 15;
    const int koffB = (lane >> 4) << 3;
    v8f acc[2];
#pragma unroll
    for (int t = 0; t < 2; ++t)
#pragma unroll
        for (int j = 0; j < 8; ++j) acc[t][j] = 0.0f;

#pragma unroll
    for (int kb = 0; kb < D1; kb += 32) {
        FragAB a;
        const bf16* pa = A + (size_t)(row0 + mA) * D1 + kb + koffA;
        a.q[0] = ld16(pa);
        a.q[1] = ld16(pa + 16);
#pragma unroll
        for (int t = 0; t < 2; ++t) {
            FragAB b;
            const bf16* pb = Bt + (size_t)(t * 16 + nB) * D1 + kb + koffB;
            b.q[0] = ld16(pb);
            b.q[1] = ld16(pb + 16);
            acc[t] = __builtin_amdgcn_wmma_f32_16x16x32_bf16(
                false, a.v, false, b.v, (short)0, acc[t], false, false);
        }
    }
    const int rbase = row0 + ((lane >> 4) << 3);
    const int nc    = lane & 15;
#pragma unroll
    for (int t = 0; t < 2; ++t)
#pragma unroll
        for (int j = 0; j < 8; ++j)
            C[(size_t)(rbase + j) * HIDC + t * 16 + nc] = acc[t][j];
}

// ---------------- attention logits per node ----------------
__global__ void k_als1(const float* __restrict__ h1, const float* __restrict__ as1,
                       const float* __restrict__ ad1, float* __restrict__ als,
                       float* __restrict__ ald) {
    int t = blockIdx.x * 256 + threadIdx.x;
    if (t >= NODES * NHEADS) return;
    int n = t >> 3, hd = t & 7;
    const float* hp = h1 + (size_t)n * D1 + hd * HIDC;
    float ss = 0.f, sd = 0.f;
#pragma unroll
    for (int c = 0; c < HIDC; ++c) {
        float hv = hp[c];
        ss += hv * as1[hd * HIDC + c];
        sd += hv * ad1[hd * HIDC + c];
    }
    als[t] = ss; ald[t] = sd;
}
__global__ void k_als2(const float* __restrict__ h2, const float* __restrict__ as2,
                       const float* __restrict__ ad2, float* __restrict__ als,
                       float* __restrict__ ald) {
    int n = blockIdx.x * 256 + threadIdx.x;
    if (n >= NODES) return;
    const float* hp = h2 + (size_t)n * HIDC;
    float ss = 0.f, sd = 0.f;
#pragma unroll
    for (int c = 0; c < HIDC; ++c) {
        float hv = hp[c];
        ss += hv * as2[c];
        sd += hv * ad2[c];
    }
    als[n] = ss; ald[n] = sd;
}

// ---------------- edge scatter (softmax numerator/denominator) ----------------
// one wave per edge; lane = channel. Softmax shift is dropped (shift-invariant).
__global__ __launch_bounds__(256) void k_edge1(const int* __restrict__ ei,
        const float* __restrict__ eattr, const float* __restrict__ lattr,
        const float* __restrict__ als, const float* __restrict__ ald,
        const float* __restrict__ ce, const float* __restrict__ h1,
        float* __restrict__ den, float* __restrict__ num) {
    int e = (blockIdx.x * 256 + threadIdx.x) >> 5;
    int lane = threadIdx.x & 31;
    if (e >= EPRIME) return;
    int s, d; float ea;
    if (e < EDGES) { s = ei[e]; d = ei[EDGES + e]; ea = eattr[e]; }
    else           { s = d = e - EDGES; ea = lattr[e - EDGES]; }
#pragma unroll
    for (int hd = 0; hd < NHEADS; ++hd) {
        float al = als[s * NHEADS + hd] + ald[d * NHEADS + hd] + ea * ce[hd];
        al = al > 0.f ? al : 0.2f * al;
        float w = __expf(al);
        if (lane == 0) atomicAdd(&den[d * NHEADS + hd], w);
        float v = h1[(size_t)s * D1 + hd * HIDC + lane] * w;
        atomicAdd(&num[(size_t)d * D1 + hd * HIDC + lane], v);
    }
}
__global__ __launch_bounds__(256) void k_edge2(const int* __restrict__ ei,
        const float* __restrict__ eattr, const float* __restrict__ lattr,
        const float* __restrict__ als, const float* __restrict__ ald,
        const float* __restrict__ ce, const float* __restrict__ h2,
        float* __restrict__ den, float* __restrict__ num) {
    int e = (blockIdx.x * 256 + threadIdx.x) >> 5;
    int lane = threadIdx.x & 31;
    if (e >= EPRIME) return;
    int s, d; float ea;
    if (e < EDGES) { s = ei[e]; d = ei[EDGES + e]; ea = eattr[e]; }
    else           { s = d = e - EDGES; ea = lattr[e - EDGES]; }
    float al = als[s] + ald[d] + ea * ce[NHEADS];
    al = al > 0.f ? al : 0.2f * al;
    float w = __expf(al);
    if (lane == 0) atomicAdd(&den[d], w);
    float v = h2[(size_t)s * HIDC + lane] * w;
    atomicAdd(&num[(size_t)d * HIDC + lane], v);
}

// ---------------- normalize + bias + BN + ReLU ----------------
__global__ void k_finish1(const float* __restrict__ num, const float* __restrict__ den,
                          const float* __restrict__ b1, const float* __restrict__ g1,
                          const float* __restrict__ be1, const float* __restrict__ m1,
                          const float* __restrict__ v1, bf16* __restrict__ h1r) {
    int t = blockIdx.x * 256 + threadIdx.x;
    if (t >= NODES * D1) return;
    int n = t >> 8, hc = t & 255, hd = hc >> 5;
    float v = num[t] / den[n * NHEADS + hd] + b1[hc];
    v = (v - m1[hc]) * (g1[hc] * rsqrtf(v1[hc] + EPSBN)) + be1[hc];
    v = fmaxf(v, 0.0f);
    h1r[t] = (bf16)v;
}
__global__ void k_finish2(const float* __restrict__ num, const float* __restrict__ den,
                          const float* __restrict__ b2, const float* __restrict__ g2,
                          const float* __restrict__ be2, const float* __restrict__ m2,
                          const float* __restrict__ v2, const int* __restrict__ batch,
                          float* __restrict__ pool, float* __restrict__ cntg) {
    int t = blockIdx.x * 256 + threadIdx.x;
    if (t >= NODES * HIDC) return;
    int n = t >> 5, c = t & 31;
    float v = num[t] / den[n] + b2[c];
    v = (v - m2[c]) * (g2[c] * rsqrtf(v2[c] + EPSBN)) + be2[c];
    v = fmaxf(v, 0.0f);
    int g = batch[n];
    atomicAdd(&pool[g * HIDC + c], v);
    if (c == 0) atomicAdd(&cntg[g], 1.0f);
}

// ---------------- mean pool + fc ----------------
__global__ void k_final(const float* __restrict__ pool, const float* __restrict__ cntg,
                        const float* __restrict__ fcW, const float* __restrict__ fcb,
                        float* __restrict__ out) {
    int g = blockIdx.x * 256 + threadIdx.x;
    if (g >= NGRAPH) return;
    float inv = 1.0f / fmaxf(cntg[g], 1.0f);
    float s = 0.f;
#pragma unroll
    for (int c = 0; c < HIDC; ++c) s += pool[g * HIDC + c] * inv * fcW[c];
    out[g] = s + fcb[0];
}

// ---------------- host launcher ----------------
extern "C" void kernel_launch(void* const* d_in, const int* in_sizes, int n_in,
                              void* d_out, int out_size, void* d_ws, size_t ws_size,
                              hipStream_t stream) {
    const float* x     = (const float*)d_in[0];
    const int*   ei    = (const int*)  d_in[1];
    const float* eattr = (const float*)d_in[2];
    const int*   batch = (const int*)  d_in[3];
    const float* W1    = (const float*)d_in[4];
    const float* We1   = (const float*)d_in[5];
    const float* as1   = (const float*)d_in[6];
    const float* ad1   = (const float*)d_in[7];
    const float* ae1   = (const float*)d_in[8];
    const float* b1    = (const float*)d_in[9];
    const float* g1    = (const float*)d_in[10];
    const float* be1   = (const float*)d_in[11];
    const float* m1    = (const float*)d_in[12];
    const float* v1    = (const float*)d_in[13];
    const float* W2    = (const float*)d_in[14];
    const float* We2   = (const float*)d_in[15];
    const float* as2   = (const float*)d_in[16];
    const float* ad2   = (const float*)d_in[17];
    const float* ae2   = (const float*)d_in[18];
    const float* b2    = (const float*)d_in[19];
    const float* g2    = (const float*)d_in[20];
    const float* be2   = (const float*)d_in[21];
    const float* m2    = (const float*)d_in[22];
    const float* v2    = (const float*)d_in[23];
    const float* fcW   = (const float*)d_in[24];
    const float* fcb   = (const float*)d_in[25];
    float* out = (float*)d_out;

    char* ws = (char*)d_ws;
    size_t off = 0;
    auto alloc = [&](size_t bytes) {
        size_t o = off;
        off += (bytes + 255) & ~(size_t)255;
        return o;
    };
    bf16*  x16   = (bf16*) (ws + alloc((size_t)MPAD * FK * 2));
    bf16*  w1t   = (bf16*) (ws + alloc((size_t)D1 * FK * 2));
    bf16*  w2t   = (bf16*) (ws + alloc((size_t)HIDC * D1 * 2));
    float* ce    = (float*)(ws + alloc(16 * 4));
    float* h1    = (float*)(ws + alloc((size_t)MPAD * D1 * 4));
    float* als1  = (float*)(ws + alloc((size_t)NODES * NHEADS * 4));
    float* ald1  = (float*)(ws + alloc((size_t)NODES * NHEADS * 4));
    float* cntin = (float*)(ws + alloc((size_t)NODES * 4));
    float* asum  = (float*)(ws + alloc((size_t)NODES * 4));
    float* lattr = (float*)(ws + alloc((size_t)NODES * 4));
    float* den1  = (float*)(ws + alloc((size_t)NODES * NHEADS * 4));
    float* num1  = (float*)(ws + alloc((size_t)NODES * D1 * 4));
    bf16*  h1r   = (bf16*) (ws + alloc((size_t)MPAD * D1 * 2));
    float* h2    = (float*)(ws + alloc((size_t)MPAD * HIDC * 4));
    float* als2  = (float*)(ws + alloc((size_t)NODES * 4));
    float* ald2  = (float*)(ws + alloc((size_t)NODES * 4));
    float* den2  = (float*)(ws + alloc((size_t)NODES * 4));
    float* num2  = (float*)(ws + alloc((size_t)NODES * HIDC * 4));
    float* pool  = (float*)(ws + alloc((size_t)NGRAPH * HIDC * 4));
    float* cntg  = (float*)(ws + alloc((size_t)NGRAPH * 4));

    // Zero everything we use (harness poisons ws; zeros needed for pads + accumulators)
    hipMemsetAsync(d_ws, 0, off, stream);

    // converters / precompute
    k_convx <<<(NODES * F_IN + 255) / 256, 256, 0, stream>>>(x, x16);
    k_convw1<<<(F_IN * D1 + 255) / 256, 256, 0, stream>>>(W1, w1t);
    k_convw2<<<(D1 * HIDC + 255) / 256, 256, 0, stream>>>(W2, w2t);
    k_ce    <<<1, 32, 0, stream>>>(We1, ae1, We2, ae2, ce);
    k_loopA <<<(EDGES + 255) / 256, 256, 0, stream>>>(ei, eattr, cntin, asum);
    k_loopB <<<(NODES + 255) / 256, 256, 0, stream>>>(cntin, asum, lattr);

    // layer 1
    k_gemm1  <<<dim3(MPAD / 128, D1 / 64), 256, 0, stream>>>(x16, w1t, h1);
    k_als1   <<<(NODES * NHEADS + 255) / 256, 256, 0, stream>>>(h1, as1, ad1, als1, ald1);
    k_edge1  <<<(EPRIME * 32 + 255) / 256, 256, 0, stream>>>(ei, eattr, lattr, als1, ald1,
                                                             ce, h1, den1, num1);
    k_finish1<<<(NODES * D1 + 255) / 256, 256, 0, stream>>>(num1, den1, b1, g1, be1, m1, v1, h1r);

    // layer 2
    k_gemm2  <<<MPAD / 128, 256, 0, stream>>>(h1r, w2t, h2);
    k_als2   <<<(NODES + 255) / 256, 256, 0, stream>>>(h2, as2, ad2, als2, ald2);
    k_edge2  <<<(EPRIME * 32 + 255) / 256, 256, 0, stream>>>(ei, eattr, lattr, als2, ald2,
                                                             ce, h2, den2, num2);
    k_finish2<<<(NODES * HIDC + 255) / 256, 256, 0, stream>>>(num2, den2, b2, g2, be2, m2, v2,
                                                              batch, pool, cntg);

    // pool + fc
    k_final<<<(NGRAPH + 255) / 256, 256, 0, stream>>>(pool, cntg, fcW, fcb, out);
}
